// Timeware_Propensity_Estimation_16750372454582
// MI455X (gfx1250) — compile-verified, matrix operand order checked by reading, not run on hardware
//
#include <hip/hip_runtime.h>
#include <hip/hip_bf16.h>
#include <math.h>

// ---------------- CDNA5 WMMA types ----------------
typedef __attribute__((ext_vector_type(16))) __bf16 v16bf;
typedef __attribute__((ext_vector_type(8)))  float  v8f;

#define HH   256
#define BB   256
#define LL   200
#define VV   50000
#define VPAD 50048          // 782 * 64: pads logits GEMM to the 64-wide block tile
#define G3H  768

__device__ __forceinline__ float sigmoidf_(float x) { return 1.0f / (1.0f + __expf(-x)); }

union AFrag { v16bf v; unsigned int u[8]; };

// Unguarded 16x32 bf16 fragment load (A-matrix layout per CDNA5 ISA 7.12.2)
// from a row-major [rows x ld] bf16 matrix at tile origin (r0, k0).
// Per lane: two contiguous 16-byte reads -> 2x global_load_b128.
__device__ __forceinline__ v16bf load_frag(const __bf16* __restrict__ base, int ld,
                                           int r0, int k0, int lane) {
    AFrag f;
    int r = r0 + (lane & 15);
    int khalf = (lane >> 4) * 8;
    const unsigned int* p = (const unsigned int*)(base + (size_t)r * ld + k0);
#pragma unroll
    for (int v = 0; v < 8; ++v) {
        int kk = ((v & 4) ? 16 : 0) + 2 * (v & 3) + khalf;   // even pair start
        f.u[v] = p[kk >> 1];                                  // 32b load = 2 bf16
    }
    return f.v;
}

#define WMMA_BF16(A_, B_, C_) \
    __builtin_amdgcn_wmma_f32_16x16x32_bf16(false, (A_), false, (B_), (short)0, (C_), false, false)

// ---------------- C = A @ W^T (+bias), bf16 -> f32 WMMA ----------------
// A: [M x K] bf16 row-major, W: [N x K] bf16 row-major, C: [M x N] f32.
// Block = 256 threads = 8 waves; each wave computes a 16x64 strip of C.
// Guard-free (M % 128 == 0, N % 64 == 0 at every call site; logits padded).
// Manually software-pipelined: next K-chunk's 5 fragments are loaded into a
// second register set before the current chunk's 4 WMMAs execute, so global
// loads overlap the matrix pipe instead of serializing on s_wait_loadcnt 0.
__global__ void gemm_bf16_wmma(const __bf16* __restrict__ A, const __bf16* __restrict__ W,
                               const float* __restrict__ bias, float* __restrict__ C,
                               int M, int N, int K) {
    int lane = threadIdx.x & 31;
    int wv   = threadIdx.x >> 5;
    int m0 = blockIdx.x * 128 + wv * 16;
    int n0 = blockIdx.y * 64;
    (void)M;

    v8f acc0 = {}, acc1 = {}, acc2 = {}, acc3 = {};

    // prologue: iteration 0 fragments
    v16bf a  = load_frag(A, K, m0,      0, lane);
    v16bf b0 = load_frag(W, K, n0 +  0, 0, lane);
    v16bf b1 = load_frag(W, K, n0 + 16, 0, lane);
    v16bf b2 = load_frag(W, K, n0 + 32, 0, lane);
    v16bf b3 = load_frag(W, K, n0 + 48, 0, lane);

    for (int k0 = 32; k0 < K; k0 += 32) {
        // issue next chunk's loads first (distinct regs -> stay in flight)
        v16bf an  = load_frag(A, K, m0,      k0, lane);
        v16bf bn0 = load_frag(W, K, n0 +  0, k0, lane);
        v16bf bn1 = load_frag(W, K, n0 + 16, k0, lane);
        v16bf bn2 = load_frag(W, K, n0 + 32, k0, lane);
        v16bf bn3 = load_frag(W, K, n0 + 48, k0, lane);
        // current chunk's WMMAs (A fragment reused 4x)
        acc0 = WMMA_BF16(a, b0, acc0);
        acc1 = WMMA_BF16(a, b1, acc1);
        acc2 = WMMA_BF16(a, b2, acc2);
        acc3 = WMMA_BF16(a, b3, acc3);
        a = an; b0 = bn0; b1 = bn1; b2 = bn2; b3 = bn3;
    }
    // epilogue
    acc0 = WMMA_BF16(a, b0, acc0);
    acc1 = WMMA_BF16(a, b1, acc1);
    acc2 = WMMA_BF16(a, b2, acc2);
    acc3 = WMMA_BF16(a, b3, acc3);

    int cl   = lane & 15;
    int rofs = (lane >> 4) ? 8 : 0;
    v8f accs[4] = { acc0, acc1, acc2, acc3 };
#pragma unroll
    for (int j = 0; j < 4; ++j) {
        int col = n0 + j * 16 + cl;
        float bv = (bias != nullptr) ? bias[col] : 0.0f;
#pragma unroll
        for (int i = 0; i < 8; ++i) {
            int row = m0 + i + rofs;
            C[(size_t)row * N + col] = accs[j][i] + bv;
        }
    }
}

// ---------------- fused recurrent step (GRU / AUGRU) ----------------
// One wave per 16x16 (batch x hidden) tile. Computes gh = h @ Whh^T for all
// three gates (24 WMMAs, K=256; h fragment feeds 3 WMMAs per chunk), adds
// precomputed gi (+bhh), applies gating, writes h_next (f32 + bf16 copy for
// the next step's A operand) and y[:, t, :]. Same double-buffer pipelining.
__global__ void rnn_step(const __bf16* __restrict__ h_bf, const float* __restrict__ h_f,
                         const float* __restrict__ gi, const __bf16* __restrict__ Whh,
                         const float* __restrict__ bhh, const float* __restrict__ att,
                         float* __restrict__ h_out_f, __bf16* __restrict__ h_out_bf,
                         float* __restrict__ y, int t, int augru) {
    int lane = threadIdx.x & 31;
    int b0 = blockIdx.x * 16;
    int j0 = blockIdx.y * 16;

    v8f acc0 = {}, acc1 = {}, acc2 = {};

    v16bf a  = load_frag(h_bf, HH, b0, 0, lane);
    v16bf br = load_frag(Whh, HH, 0 * HH + j0, 0, lane);
    v16bf bz = load_frag(Whh, HH, 1 * HH + j0, 0, lane);
    v16bf bn = load_frag(Whh, HH, 2 * HH + j0, 0, lane);

    for (int k0 = 32; k0 < HH; k0 += 32) {
        v16bf an  = load_frag(h_bf, HH, b0, k0, lane);
        v16bf brn = load_frag(Whh, HH, 0 * HH + j0, k0, lane);
        v16bf bzn = load_frag(Whh, HH, 1 * HH + j0, k0, lane);
        v16bf bnn = load_frag(Whh, HH, 2 * HH + j0, k0, lane);
        acc0 = WMMA_BF16(a, br, acc0);
        acc1 = WMMA_BF16(a, bz, acc1);
        acc2 = WMMA_BF16(a, bn, acc2);
        a = an; br = brn; bz = bzn; bn = bnn;
    }
    acc0 = WMMA_BF16(a, br, acc0);
    acc1 = WMMA_BF16(a, bz, acc1);
    acc2 = WMMA_BF16(a, bn, acc2);

    int col = j0 + (lane & 15);
    float bh_r = 0.0f, bh_z = 0.0f, bh_n = 0.0f;
    if (bhh != nullptr) {
        bh_r = bhh[col]; bh_z = bhh[HH + col]; bh_n = bhh[2 * HH + col];
    }
#pragma unroll
    for (int i = 0; i < 8; ++i) {
        int b = b0 + i + ((lane >> 4) ? 8 : 0);
        size_t gbase = ((size_t)b * LL + t) * (size_t)G3H;
        float gir = gi[gbase + col];
        float giz = gi[gbase + HH + col];
        float gin = gi[gbase + 2 * HH + col];
        float ghr = acc0[i] + bh_r;
        float ghz = acc1[i] + bh_z;
        float ghn = acc2[i] + bh_n;
        float hold = h_f[(size_t)b * HH + col];
        float r  = sigmoidf_(gir + ghr);
        float nn = tanhf(gin + r * ghn);
        float hnew;
        if (augru) {
            float u = sigmoidf_(giz + ghz) * att[(size_t)b * LL + t];
            hnew = (1.0f - u) * hold + u * nn;
        } else {
            float z = sigmoidf_(giz + ghz);
            hnew = (1.0f - z) * nn + z * hold;
        }
        h_out_f [(size_t)b * HH + col] = hnew;
        h_out_bf[(size_t)b * HH + col] = (__bf16)hnew;
        y[((size_t)b * LL + t) * HH + col] = hnew;
    }
}

// ---------------- small helpers ----------------
__global__ void cast_bf16(const float* __restrict__ src, __bf16* __restrict__ dst, size_t n) {
    size_t i = (size_t)blockIdx.x * blockDim.x + threadIdx.x;
    if (i < n) dst[i] = (__bf16)src[i];
}

__global__ void zero_h(float* __restrict__ f, __bf16* __restrict__ b) {
    int i = blockIdx.x * blockDim.x + threadIdx.x;
    if (i < BB * HH) { f[i] = 0.0f; b[i] = (__bf16)0.0f; }
}

__global__ void gather_last(const float* __restrict__ seq, const int* __restrict__ seq_len,
                            float* __restrict__ out_f, __bf16* __restrict__ out_bf) {
    int b = blockIdx.x, h = threadIdx.x;
    int idx = seq_len[b] - 1;
    if (idx < 0) idx = 0;
    if (idx >= LL) idx = LL - 1;
    float v = seq[((size_t)b * LL + idx) * HH + h];
    if (out_f)  out_f [(size_t)b * HH + h] = v;
    if (out_bf) out_bf[(size_t)b * HH + h] = (__bf16)v;
}

// logits laid out with padded row stride (VPAD); only first VV columns are real.
__global__ void nll_row(const float* __restrict__ logits, const int* __restrict__ target,
                        float* __restrict__ row_loss) {
    __shared__ float sm[256];
    int b = blockIdx.x, tid = threadIdx.x;
    const float* row = logits + (size_t)b * VPAD;
    float mx = -3.4e38f;
    for (int i = tid; i < VV; i += 256) mx = fmaxf(mx, row[i]);
    sm[tid] = mx; __syncthreads();
    for (int s = 128; s > 0; s >>= 1) { if (tid < s) sm[tid] = fmaxf(sm[tid], sm[tid + s]); __syncthreads(); }
    mx = sm[0]; __syncthreads();
    float sum = 0.0f;
    for (int i = tid; i < VV; i += 256) sum += __expf(row[i] - mx);
    sm[tid] = sum; __syncthreads();
    for (int s = 128; s > 0; s >>= 1) { if (tid < s) sm[tid] += sm[tid + s]; __syncthreads(); }
    if (tid == 0) {
        float lse = mx + logf(sm[0]);
        row_loss[b] = -(row[target[b]] - lse);
    }
}

__global__ void mean_reduce(const float* __restrict__ row_loss, float* __restrict__ out) {
    __shared__ float sm[256];
    int tid = threadIdx.x;
    sm[tid] = row_loss[tid]; __syncthreads();
    for (int s = 128; s > 0; s >>= 1) { if (tid < s) sm[tid] += sm[tid + s]; __syncthreads(); }
    if (tid == 0) out[0] = sm[0] / 256.0f;
}

// ---------------- host orchestration ----------------
extern "C" void kernel_launch(void* const* d_in, const int* in_sizes, int n_in,
                              void* d_out, int out_size, void* d_ws, size_t ws_size,
                              hipStream_t stream) {
    (void)in_sizes; (void)n_in; (void)out_size; (void)ws_size;
    const float* item_seq_emb = (const float*)d_in[0];
    const int*   target_item  = (const int*)d_in[1];
    const float* item_emb     = (const float*)d_in[2];
    const int*   seq_len      = (const int*)d_in[3];
    /* d_in[4] target_item_feat_emb unused by the reference */
    const float* tr_output    = (const float*)d_in[5];
    const float* W_ih0 = (const float*)d_in[6];
    const float* W_hh0 = (const float*)d_in[7];
    const float* W_ih1 = (const float*)d_in[8];
    const float* W_hh1 = (const float*)d_in[9];
    const float* W_d   = (const float*)d_in[10];
    const float* b_d   = (const float*)d_in[11];
    const float* A_Wih = (const float*)d_in[12];
    const float* A_Whh = (const float*)d_in[13];
    const float* A_bih = (const float*)d_in[14];
    const float* A_bhh = (const float*)d_in[15];
    float* out = (float*)d_out;

    // ---- workspace carve-out ----
    char* ws = (char*)d_ws;
    auto carve = [&](size_t bytes) -> char* {
        char* p = ws; ws += (bytes + 255) & ~(size_t)255; return p;
    };
    float*  gi     = (float*) carve((size_t)BB * LL * G3H * 4);   // 157 MB
    // x_bf also holds the bf16 item_emb padded to VPAD rows:
    // VPAD*HH = 12,812,288 elems <= BB*LL*HH = 13,107,200 elems.
    __bf16* x_bf   = (__bf16*)carve((size_t)BB * LL * HH * 2);    // 26 MB
    float*  seqA   = (float*) carve((size_t)BB * LL * HH * 4);    // 52 MB
    float*  seqB   = (float*) carve((size_t)BB * LL * HH * 4);    // 52 MB
    __bf16* wih0b  = (__bf16*)carve((size_t)G3H * HH * 2);
    __bf16* whh0b  = (__bf16*)carve((size_t)G3H * HH * 2);
    __bf16* wih1b  = (__bf16*)carve((size_t)G3H * HH * 2);
    __bf16* whh1b  = (__bf16*)carve((size_t)G3H * HH * 2);
    __bf16* awihb  = (__bf16*)carve((size_t)G3H * HH * 2);
    __bf16* awhhb  = (__bf16*)carve((size_t)G3H * HH * 2);
    __bf16* wdb    = (__bf16*)carve((size_t)HH * HH * 2);
    float*  hAf    = (float*) carve((size_t)BB * HH * 4);
    float*  hBf    = (float*) carve((size_t)BB * HH * 4);
    __bf16* hAb    = (__bf16*)carve((size_t)BB * HH * 2);
    __bf16* hBb    = (__bf16*)carve((size_t)BB * HH * 2);
    __bf16* seqout_bf = (__bf16*)carve((size_t)BB * HH * 2);
    float*  logits = (float*) carve((size_t)BB * VPAD * 4);       // 51 MB (padded stride)
    float*  row_loss = (float*)carve((size_t)BB * 4);

    auto castN = [&](const float* s, __bf16* d, size_t n) {
        cast_bf16<<<dim3((unsigned)((n + 255) / 256)), dim3(256), 0, stream>>>(s, d, n);
    };

    // bf16 weight copies
    castN(W_ih0, wih0b, (size_t)G3H * HH);
    castN(W_hh0, whh0b, (size_t)G3H * HH);
    castN(W_ih1, wih1b, (size_t)G3H * HH);
    castN(W_hh1, whh1b, (size_t)G3H * HH);
    castN(A_Wih, awihb, (size_t)G3H * HH);
    castN(A_Whh, awhhb, (size_t)G3H * HH);
    castN(W_d,   wdb,   (size_t)HH * HH);

    // M % 128 == 0 and N % 64 == 0 for every call site (logits padded to VPAD).
    auto gemm = [&](const __bf16* A, const __bf16* W, const float* bias, float* C,
                    int M, int N, int K) {
        gemm_bf16_wmma<<<dim3((unsigned)(M / 128), (unsigned)(N / 64)),
                         dim3(256), 0, stream>>>(A, W, bias, C, M, N, K);
    };

    auto run_recurrence = [&](const __bf16* xb, const __bf16* wihb, const __bf16* whhb,
                              const float* bih, const float* bhh, const float* att,
                              float* yout, int augru) {
        // batched input projection: gi = x @ Wih^T (+ bih)   [B*L, 3H]
        gemm(xb, wihb, bih, gi, BB * LL, G3H, HH);
        zero_h<<<dim3((BB * HH) / 256), dim3(256), 0, stream>>>(hAf, hAb);
        float* hf_in = hAf; __bf16* hb_in = hAb;
        float* hf_out = hBf; __bf16* hb_out = hBb;
        for (int t = 0; t < LL; ++t) {
            rnn_step<<<dim3(BB / 16, HH / 16), dim3(32), 0, stream>>>(
                hb_in, hf_in, gi, whhb, bhh, att, hf_out, hb_out, yout, t, augru);
            float* tf = hf_in; hf_in = hf_out; hf_out = tf;
            __bf16* tb = hb_in; hb_in = hb_out; hb_out = tb;
        }
    };

    // GRU layer 0: item_seq_emb -> seqA
    castN(item_seq_emb, x_bf, (size_t)BB * LL * HH);
    run_recurrence(x_bf, wih0b, whh0b, nullptr, nullptr, nullptr, seqA, 0);

    // GRU layer 1: seqA -> seqB
    castN(seqA, x_bf, (size_t)BB * LL * HH);
    run_recurrence(x_bf, wih1b, whh1b, nullptr, nullptr, nullptr, seqB, 0);

    // dense: gru_output = seqB @ W_d^T + b_d -> seqA
    castN(seqB, x_bf, (size_t)BB * LL * HH);
    gemm(x_bf, wdb, b_d, seqA, BB * LL, HH, HH);

    // seq_output (last valid step of gru_output), bf16 for logits GEMM
    gather_last<<<dim3(BB), dim3(HH), 0, stream>>>(seqA, seq_len, nullptr, seqout_bf);

    // AUGRU over gru_output with attention tr_output -> seqB
    castN(seqA, x_bf, (size_t)BB * LL * HH);
    run_recurrence(x_bf, awihb, awhhb, A_bih, A_bhh, tr_output, seqB, 1);

    // evolution = last valid step of AUGRU output -> out[1..]
    gather_last<<<dim3(BB), dim3(HH), 0, stream>>>(seqB, seq_len, out + 1, nullptr);

    // logits = seq_output @ item_emb^T, padded to VPAD columns (rows of x_bf
    // beyond VV read leftover in-bounds data; padded logit columns never read).
    castN(item_emb, x_bf, (size_t)VV * HH);
    gemm(seqout_bf, x_bf, nullptr, logits, BB, VPAD, HH);

    // aux_loss = -mean(log_softmax(logits)[b, target[b]]) -> out[0]
    nll_row<<<dim3(BB), dim3(256), 0, stream>>>(logits, target_item, row_loss);
    mean_reduce<<<dim3(1), dim3(256), 0, stream>>>(row_loss, out);
}
